// Qwen3OmniMoeThinkerTextSparseMoeBlock_41446434406896
// MI455X (gfx1250) — compile-verified, matrix operand order checked by reading, not run on hardware
//
#include <hip/hip_runtime.h>
#include <hip/hip_bf16.h>
#include <stdint.h>

// Problem constants (match reference)
#define NT 2048      // tokens (B*S)
#define NH 2048      // hidden
#define NI 768       // intermediate
#define NE 64        // experts
#define NK 8         // top-k
#define MAX_ROWS 17408               // 16384 assignments + per-expert pad to 16
#define MAX_TILES (MAX_ROWS / 16)    // 1088

typedef __bf16 bf16;
typedef __attribute__((ext_vector_type(16))) __bf16 v16bf;
typedef __attribute__((ext_vector_type(8)))  float  v8f;

// 16-bit WMMA operand fragment (16 bf16 = 8 VGPRs per lane)
union Frag16 {
    v16bf v;
    bf16  h[16];
    uint4 q[2];
};

// ---------------------------------------------------------------------------
// Init: zero per-expert counters, poison row->token map with -1 (pad marker)
// ---------------------------------------------------------------------------
__global__ void init_kernel(int* __restrict__ cnt, int* __restrict__ fill,
                            int* __restrict__ rows_token) {
    int i = blockIdx.x * blockDim.x + threadIdx.x;
    if (i < NE) { cnt[i] = 0; fill[i] = 0; }
    if (i < MAX_ROWS) rows_token[i] = -1;
}

// ---------------------------------------------------------------------------
// Convert activations to bf16 once; row NT (one extra row) is all zeros and
// serves as the branch-free A-matrix source for pad rows.
// ---------------------------------------------------------------------------
__global__ void convert_x_kernel(const float* __restrict__ x, bf16* __restrict__ xb) {
    int i = blockIdx.x * blockDim.x + threadIdx.x;
    if (i < (NT + 1) * NH) xb[i] = (i < NT * NH) ? (bf16)x[i] : (bf16)0.f;
}

// ---------------------------------------------------------------------------
// Router: logits = x @ w_router, softmax, top-8, renormalize.
// One block (64 threads, 2 waves) per token; x row staged in LDS.
// ---------------------------------------------------------------------------
__global__ __launch_bounds__(64)
void router_kernel(const float* __restrict__ x, const float* __restrict__ wr,
                   float* __restrict__ logits, int* __restrict__ topk_e,
                   float* __restrict__ topk_w, int* __restrict__ cnt) {
    __shared__ float xs[NH];
    __shared__ float pr[NE];
    const int t = blockIdx.x;
    const float* xrow = x + (size_t)t * NH;
    for (int i = threadIdx.x; i < NH; i += 64) xs[i] = xrow[i];
    __syncthreads();

    const int e = threadIdx.x;     // one expert per thread (NE == 64)
    float acc = 0.f;
#pragma unroll 8
    for (int h = 0; h < NH; ++h) acc = fmaf(xs[h], wr[(size_t)h * NE + e], acc);
    logits[(size_t)t * NE + e] = acc;   // raw router logits (2nd output)
    pr[e] = acc;
    __syncthreads();

    if (e == 0) {
        float mx = pr[0];
        for (int i = 1; i < NE; ++i) mx = fmaxf(mx, pr[i]);
        float s = 0.f;
        for (int i = 0; i < NE; ++i) { pr[i] = __expf(pr[i] - mx); s += pr[i]; }
        const float inv = 1.f / s;
        int sel[NK]; float sw[NK]; float wsum = 0.f;
        for (int k = 0; k < NK; ++k) {
            int bi = 0; float bv = -1.f;
            for (int i = 0; i < NE; ++i)
                if (pr[i] > bv) { bv = pr[i]; bi = i; }
            sel[k] = bi; sw[k] = bv * inv; wsum += sw[k]; pr[bi] = -1.f;
        }
        const float norm = 1.f / fmaxf(wsum, 1e-9f);
        for (int k = 0; k < NK; ++k) {
            topk_e[t * NK + k] = sel[k];
            topk_w[t * NK + k] = sw[k] * norm;
            atomicAdd(&cnt[sel[k]], 1);
        }
    }
}

// ---------------------------------------------------------------------------
// Scan: exclusive prefix of counts, padded to multiples of 16 rows so each
// WMMA M-tile maps to exactly one expert; build tile -> expert map.
// ---------------------------------------------------------------------------
__global__ void scan_kernel(const int* __restrict__ cnt, int* __restrict__ poff,
                            int* __restrict__ tile_expert) {
    if (blockIdx.x != 0 || threadIdx.x != 0) return;
    int off = 0;
    for (int e = 0; e < NE; ++e) {
        poff[e] = off;
        off += (cnt[e] + 15) & ~15;
    }
    poff[NE] = off;
    for (int tl = 0; tl < MAX_TILES; ++tl) tile_expert[tl] = -1;
    for (int e = 0; e < NE; ++e) {
        const int t0 = poff[e] >> 4;
        const int t1 = (poff[e] + ((cnt[e] + 15) & ~15)) >> 4;
        for (int tl = t0; tl < t1; ++tl) tile_expert[tl] = e;
    }
}

// ---------------------------------------------------------------------------
// Fill: scatter (token, weight) into the compacted, expert-grouped row space.
// ---------------------------------------------------------------------------
__global__ void fill_kernel(const int* __restrict__ topk_e, const float* __restrict__ topk_w,
                            const int* __restrict__ poff, int* __restrict__ fill,
                            int* __restrict__ rows_token, float* __restrict__ rows_w) {
    const int idx = blockIdx.x * blockDim.x + threadIdx.x;
    if (idx >= NT * NK) return;
    const int e = topk_e[idx];
    const int pos = atomicAdd(&fill[e], 1);
    const int row = poff[e] + pos;
    rows_token[row] = idx >> 3;      // token index
    rows_w[row]     = topk_w[idx];
}

// ---------------------------------------------------------------------------
// MLP stage 1: h = silu(Xe @ Wg) * (Xe @ Wu), bf16 WMMA, f32 accumulate.
// grid.x = M-tile (16 rows), grid.y = 64-col chunk of I; 4 waves per block,
// one 16x16 WMMA tile per wave. Pad rows read the all-zero xb row (branch-
// free inner loop, EXEC stays all-ones for WMMA).
// ---------------------------------------------------------------------------
__global__ __launch_bounds__(128)
void mlp1_kernel(const bf16* __restrict__ xb,
                 const float* __restrict__ wg, const float* __restrict__ wu,
                 const int* __restrict__ rows_token,
                 const int* __restrict__ tile_expert,
                 bf16* __restrict__ hbuf) {
    const int tile = blockIdx.x;
    const int e = tile_expert[tile];
    if (e < 0) return;                        // uniform per block
    const int lane = threadIdx.x & 31;
    const int wave = threadIdx.x >> 5;
    const int n0   = blockIdx.y * 64 + wave * 16;
    const int col  = lane & 15;               // N column / A row (per layout)
    const int kb   = (lane >> 4) * 8;         // K-half select per 16-bit layout
    const int arow = (tile << 4) + col;
    const int token = rows_token[arow];       // -1 for pad rows
    const int tokld = (token >= 0) ? token : NT;   // pad -> zero row, no branch
    const bf16* abase = xb + (size_t)tokld * NH + kb;
    const float* wgp = wg + (size_t)e * NH * NI;
    const float* wup = wu + (size_t)e * NH * NI;

    v8f accG = {}; v8f accU = {};
    for (int k0 = 0; k0 < NH; k0 += 32) {
        Frag16 a;
        a.q[0] = *reinterpret_cast<const uint4*>(abase + k0);        // K = k0+kb..+7
        a.q[1] = *reinterpret_cast<const uint4*>(abase + k0 + 16);   // K = k0+16+kb..+7
        Frag16 bg, bu;
        const float* wgk = wgp + (size_t)(k0 + kb) * NI + n0 + col;
        const float* wuk = wup + (size_t)(k0 + kb) * NI + n0 + col;
        if (k0 + 64 <= NH) {                  // stream next weight panel
            __builtin_prefetch(wgk + (size_t)32 * NI, 0, 1);
            __builtin_prefetch(wuk + (size_t)32 * NI, 0, 1);
        }
#pragma unroll
        for (int i = 0; i < 8; ++i) {
            bg.h[i]     = (bf16)wgk[(size_t)i * NI];
            bg.h[i + 8] = (bf16)wgk[(size_t)(i + 16) * NI];
            bu.h[i]     = (bf16)wuk[(size_t)i * NI];
            bu.h[i + 8] = (bf16)wuk[(size_t)(i + 16) * NI];
        }
        accG = __builtin_amdgcn_wmma_f32_16x16x32_bf16(false, a.v, false, bg.v,
                                                       (short)0, accG, false, false);
        accU = __builtin_amdgcn_wmma_f32_16x16x32_bf16(false, a.v, false, bu.v,
                                                       (short)0, accU, false, false);
    }

    const int g4 = lane >> 4;
#pragma unroll
    for (int r = 0; r < 8; ++r) {
        const int row = (tile << 4) + r + 8 * g4;      // C/D layout: VGPR r -> M=r(+8)
        const float gv = accG[r];
        const float uv = accU[r];
        // silu(g)*u with v_rcp_f32 instead of the IEEE division ladder
        const float hv = gv * uv * __builtin_amdgcn_rcpf(1.0f + __expf(-gv));
        hbuf[(size_t)row * NI + n0 + col] = (bf16)hv;
    }
}

// ---------------------------------------------------------------------------
// MLP stage 2: Y = h @ Wd, scaled by router weight, scatter-add into output.
// ---------------------------------------------------------------------------
__global__ __launch_bounds__(128)
void mlp2_kernel(const bf16* __restrict__ hbuf,
                 const float* __restrict__ wd,
                 const int* __restrict__ rows_token,
                 const float* __restrict__ rows_w,
                 const int* __restrict__ tile_expert,
                 float* __restrict__ out) {
    const int tile = blockIdx.x;
    const int e = tile_expert[tile];
    if (e < 0) return;
    const int lane = threadIdx.x & 31;
    const int wave = threadIdx.x >> 5;
    const int n0   = blockIdx.y * 64 + wave * 16;
    const int col  = lane & 15;
    const int kb   = (lane >> 4) * 8;
    const int arow = (tile << 4) + col;
    const bf16* abase = hbuf + (size_t)arow * NI + kb;
    const float* wdp = wd + (size_t)e * NI * NH;

    v8f acc = {};
    for (int k0 = 0; k0 < NI; k0 += 32) {
        Frag16 a;
        a.q[0] = *reinterpret_cast<const uint4*>(abase + k0);
        a.q[1] = *reinterpret_cast<const uint4*>(abase + k0 + 16);
        Frag16 b;
        const float* wdk = wdp + (size_t)(k0 + kb) * NH + n0 + col;
        if (k0 + 64 <= NI) {
            __builtin_prefetch(wdk + (size_t)32 * NH, 0, 1);
        }
#pragma unroll
        for (int i = 0; i < 8; ++i) {
            b.h[i]     = (bf16)wdk[(size_t)i * NH];
            b.h[i + 8] = (bf16)wdk[(size_t)(i + 16) * NH];
        }
        acc = __builtin_amdgcn_wmma_f32_16x16x32_bf16(false, a.v, false, b.v,
                                                      (short)0, acc, false, false);
    }

    const int g4 = lane >> 4;
#pragma unroll
    for (int r = 0; r < 8; ++r) {
        const int row = (tile << 4) + r + 8 * g4;
        const int token = rows_token[row];
        if (token >= 0) {
            atomicAdd(&out[(size_t)token * NH + n0 + col], acc[r] * rows_w[row]);
        }
    }
}

// ---------------------------------------------------------------------------
// Host launcher
// ---------------------------------------------------------------------------
extern "C" void kernel_launch(void* const* d_in, const int* in_sizes, int n_in,
                              void* d_out, int out_size, void* d_ws, size_t ws_size,
                              hipStream_t stream) {
    (void)in_sizes; (void)n_in; (void)out_size;
    const float* x  = (const float*)d_in[0];   // [1, 2048, 2048]
    const float* wr = (const float*)d_in[1];   // [2048, 64]
    const float* wg = (const float*)d_in[2];   // [64, 2048, 768]
    const float* wu = (const float*)d_in[3];   // [64, 2048, 768]
    const float* wd = (const float*)d_in[4];   // [64, 768, 2048]

    float* out_final  = (float*)d_out;                    // [NT, NH]
    float* out_logits = out_final + (size_t)NT * NH;      // [NT, NE]

    // Workspace carve-out (~35.5 MB total, 256B-aligned slabs)
    char* ws = (char*)d_ws;
    size_t off = 0;
    auto take = [&](size_t bytes) {
        size_t o = off;
        off = (off + bytes + 255) & ~(size_t)255;
        return o;
    };
    int*   cnt         = (int*)  (ws + take(NE * 4));
    int*   fill        = (int*)  (ws + take(NE * 4));
    int*   poff        = (int*)  (ws + take((NE + 1) * 4));
    int*   tile_expert = (int*)  (ws + take(MAX_TILES * 4));
    int*   topk_e      = (int*)  (ws + take((size_t)NT * NK * 4));
    float* topk_w      = (float*)(ws + take((size_t)NT * NK * 4));
    int*   rows_token  = (int*)  (ws + take((size_t)MAX_ROWS * 4));
    float* rows_w      = (float*)(ws + take((size_t)MAX_ROWS * 4));
    bf16*  xb          = (bf16*) (ws + take((size_t)(NT + 1) * NH * 2)); // +1 zero row
    bf16*  hbuf        = (bf16*) (ws + take((size_t)MAX_ROWS * NI * 2));
    (void)ws_size;

    // Output accumulator must start at zero (scatter-add epilogue)
    hipMemsetAsync(out_final, 0, (size_t)NT * NH * sizeof(float), stream);

    init_kernel<<<(MAX_ROWS + 255) / 256, 256, 0, stream>>>(cnt, fill, rows_token);
    convert_x_kernel<<<((NT + 1) * NH + 255) / 256, 256, 0, stream>>>(x, xb);
    router_kernel<<<NT, 64, 0, stream>>>(x, wr, out_logits, topk_e, topk_w, cnt);
    scan_kernel<<<1, 1, 0, stream>>>(cnt, poff, tile_expert);
    fill_kernel<<<(NT * NK + 255) / 256, 256, 0, stream>>>(topk_e, topk_w, poff, fill,
                                                           rows_token, rows_w);

    dim3 blk(128);
    dim3 g1(MAX_TILES, NI / 64);   // 1088 x 12
    mlp1_kernel<<<g1, blk, 0, stream>>>(xb, wg, wu, rows_token, tile_expert, hbuf);

    dim3 g2(MAX_TILES, NH / 64);   // 1088 x 32
    mlp2_kernel<<<g2, blk, 0, stream>>>(hbuf, wd, rows_token, rows_w, tile_expert,
                                        out_final);
}
